// TFN_16887811407922
// MI455X (gfx1250) — matrix operand order repaired and checked by reading, not compiled, not run on hardware
//
#include <hip/hip_runtime.h>
#include <hip/hip_bf16.h>

// ---------------------------------------------------------------------------
// TFN inference for MI455X (gfx1250, wave32, WMMA f16 16x16x32, f32 accum)
// ---------------------------------------------------------------------------

#define S_DIM 128
#define B_DIM 128
#define N_ROWS (S_DIM * B_DIM)        // 16384
#define TEXT_IN 768
#define AUDIO_IN 128
#define TEXT_H 128
#define AUDIO_H 32
#define A_COLS 33                     // 1 + AUDIO_H
#define A_PAD 34                      // LDS row pad for scale tile
#define T_COLS 129                    // 1 + TEXT_H
#define T_PAD 160                     // 129 padded to multiple of 32
#define POST 768
#define EPSV 1e-5f

typedef __attribute__((ext_vector_type(16))) _Float16 v16h;
typedef __attribute__((ext_vector_type(8)))  _Float16 v8h;
typedef __attribute__((ext_vector_type(8)))  float    v8f;
typedef __attribute__((ext_vector_type(4)))  float    v4f;

__device__ __forceinline__ v8f wmma_f16(v16h a, v16h b, v8f c) {
  // D = A(16x32,f16) x B(32x16,f16) + C(16x16,f32)
  return __builtin_amdgcn_wmma_f32_16x16x32_f16(
      /*neg_a=*/false, a, /*neg_b=*/false, b,
      /*c_mod=*/(short)0, c, /*reuse_a=*/false, /*reuse_b=*/false);
}

// A/B fragment from two 16B chunks of f16 (p0: elems 0..7, p1: elems 8..15)
__device__ __forceinline__ v16h frag16(const _Float16* p0, const _Float16* p1) {
  union { v16h v; v8h h[2]; } u;
  u.h[0] = *(const v8h*)p0;
  u.h[1] = *(const v8h*)p1;
  return u.v;
}

// A fragment from f32 source; chunk0 at p0[0..7], chunk1 at p0[16..23]
__device__ __forceinline__ v16h frag16_from_f32(const float* p0) {
  v4f a = *(const v4f*)(p0);
  v4f b = *(const v4f*)(p0 + 4);
  v4f c = *(const v4f*)(p0 + 16);
  v4f d = *(const v4f*)(p0 + 20);
  v16h r;
#pragma unroll
  for (int i = 0; i < 4; ++i) {
    r[i]      = (_Float16)a[i];
    r[4 + i]  = (_Float16)b[i];
    r[8 + i]  = (_Float16)c[i];
    r[12 + i] = (_Float16)d[i];
  }
  return r;
}

// ---------------------------------------------------------------------------
// K0: fold BN into weights; build f16, transposed / K-padded weight slabs.
//   W1t[a][p][k] : [33][768][160] f16, k<129 -> W1[a*129+k][p], else 0
//   W2t[p][k]    : [768][768]     f16, = W2[k][p]
//   tWeT[p][k]   : [128][768]     f16, = sT[k]*tW[k][p]
//   aWeT[p][k]   : [32][128]      f16, = sA[k]*aW[k][p]
//   biasT[p], biasA[p] : BN shift folded into linear bias (f32)
// ---------------------------------------------------------------------------
#define NW1 (33LL * 768 * 160)
#define NW2 (768LL * 768)
#define NTW (128LL * 768)
#define NAW (32LL * 128)
#define K0_TOTAL (NW1 + NW2 + NTW + NAW + 128 + 32)

__global__ __launch_bounds__(256) void k_prep(
    const float* __restrict__ aW, const float* __restrict__ ab,
    const float* __restrict__ a_gamma, const float* __restrict__ a_beta,
    const float* __restrict__ a_mean, const float* __restrict__ a_var,
    const float* __restrict__ tW, const float* __restrict__ tb,
    const float* __restrict__ t_gamma, const float* __restrict__ t_beta,
    const float* __restrict__ t_mean, const float* __restrict__ t_var,
    const float* __restrict__ W1, const float* __restrict__ W2,
    _Float16* __restrict__ W1t, _Float16* __restrict__ W2t,
    _Float16* __restrict__ tWeT, _Float16* __restrict__ aWeT,
    float* __restrict__ biasT, float* __restrict__ biasA) {
  long long idx = (long long)blockIdx.x * 256 + threadIdx.x;
  if (idx >= K0_TOTAL) return;
  if (idx < NW1) {
    long long a = idx / (768LL * 160);
    long long rem = idx % (768LL * 160);
    long long p = rem / 160, k = rem % 160;
    float v = (k < 129) ? W1[(a * 129 + k) * 768 + p] : 0.0f;
    W1t[idx] = (_Float16)v;
    return;
  }
  idx -= NW1;
  if (idx < NW2) {
    long long p = idx / 768, k = idx % 768;
    W2t[idx] = (_Float16)W2[k * 768 + p];
    return;
  }
  idx -= NW2;
  if (idx < NTW) {
    long long p = idx / 768, k = idx % 768;
    float s = t_gamma[k] * rsqrtf(t_var[k] + EPSV);
    tWeT[idx] = (_Float16)(tW[k * 128 + p] * s);
    return;
  }
  idx -= NTW;
  if (idx < NAW) {
    long long p = idx / 128, k = idx % 128;
    float s = a_gamma[k] * rsqrtf(a_var[k] + EPSV);
    aWeT[idx] = (_Float16)(aW[k * 32 + p] * s);
    return;
  }
  idx -= NAW;
  if (idx < 128) {
    int p = (int)idx;
    float acc = tb[p];
    for (int k = 0; k < 768; ++k) {
      float s = t_gamma[k] * rsqrtf(t_var[k] + EPSV);
      acc += (t_beta[k] - t_mean[k] * s) * tW[k * 128 + p];
    }
    biasT[p] = acc;
    return;
  }
  idx -= 128;
  {
    int p = (int)idx;
    float acc = ab[p];
    for (int k = 0; k < 128; ++k) {
      float s = a_gamma[k] * rsqrtf(a_var[k] + EPSV);
      acc += (a_beta[k] - a_mean[k] * s) * aW[k * 32 + p];
    }
    biasA[p] = acc;
  }
}

// ---------------------------------------------------------------------------
// K1: branch GEMMs via WMMA.
//   _t (f16 [N,160]): col0 = 1, cols 1..128 = text_h, cols 129..159 = 0
//   _a (f32 [N,33]) : col0 = 1, cols 1..32  = audio_h
// One workgroup per 128 rows. 8 waves; wave w owns row-tile w (16 rows).
// ---------------------------------------------------------------------------
__global__ __launch_bounds__(256) void k_branches(
    const float* __restrict__ text_x, const float* __restrict__ audio_x,
    const _Float16* __restrict__ tWeT, const float* __restrict__ biasT,
    const _Float16* __restrict__ aWeT, const float* __restrict__ biasA,
    _Float16* __restrict__ t_out, float* __restrict__ a_out) {
  const int r0 = blockIdx.x * 128;
  const int tid = threadIdx.x;
  const int w = tid >> 5, lane = tid & 31;
  const int ln = lane & 15, half = lane >> 4;

  // Padding / constant columns (disjoint from GEMM outputs -> no ordering hazard)
  if (tid < 128) {
    const long long n = r0 + tid;
    t_out[n * T_PAD + 0] = (_Float16)1.0f;
#pragma unroll
    for (int k = T_COLS; k < T_PAD; ++k) t_out[n * T_PAD + k] = (_Float16)0.0f;
    a_out[n * A_COLS + 0] = 1.0f;
  }

  const int rowm = r0 + w * 16 + ln;  // A-operand row for this lane

  // ---- audio branch: [128,128] x [128,32] ----
  {
    v8f acc[2] = {};
#pragma unroll
    for (int ks = 0; ks < 4; ++ks) {
      const float* ap = audio_x + (long long)rowm * AUDIO_IN + ks * 32 + 8 * half;
      v16h af = frag16_from_f32(ap);
#pragma unroll
      for (int ct = 0; ct < 2; ++ct) {
        const int p = ct * 16 + ln;
        const _Float16* bp = aWeT + (long long)p * AUDIO_IN + ks * 32 + 8 * half;
        acc[ct] = wmma_f16(af, frag16(bp, bp + 16), acc[ct]);
      }
    }
#pragma unroll
    for (int ct = 0; ct < 2; ++ct) {
      const int p = ct * 16 + ln;
      const float bias = biasA[p];
#pragma unroll
      for (int r = 0; r < 8; ++r) {
        const long long n = r0 + w * 16 + half * 8 + r;
        a_out[n * A_COLS + 1 + p] = acc[ct][r] + bias;
      }
    }
  }

  // ---- text branch: [128,768] x [768,128] ----
  {
    v8f acc[8] = {};
    for (int ks = 0; ks < 24; ++ks) {
      const float* ap = text_x + (long long)rowm * TEXT_IN + ks * 32 + 8 * half;
      v16h af = frag16_from_f32(ap);
#pragma unroll
      for (int ct = 0; ct < 8; ++ct) {
        const int p = ct * 16 + ln;
        const _Float16* bp = tWeT + (long long)p * TEXT_IN + ks * 32 + 8 * half;
        acc[ct] = wmma_f16(af, frag16(bp, bp + 16), acc[ct]);
      }
    }
#pragma unroll
    for (int ct = 0; ct < 8; ++ct) {
      const int p = ct * 16 + ln;
      const float bias = biasT[p];
#pragma unroll
      for (int r = 0; r < 8; ++r) {
        const long long n = r0 + w * 16 + half * 8 + r;
        t_out[n * T_PAD + 1 + p] = (_Float16)(acc[ct][r] + bias);
      }
    }
  }
}

// ---------------------------------------------------------------------------
// K2: y1 = relu(fusion @ W1 + b1), fusion never materialized anywhere.
// A-fragment rows are per-lane (lane l holds row l%16 only), so
// diag(_a[:,a]) is applied as a per-lane scalar f16 multiply on the raw _t
// fragment in registers. LDS holds the _t tile + f16 scale tile, staged ONCE.
// grid = (3, 128): x = 256-col block, y = 128-row block. 8 waves:
//   wave -> (row group w&3: 32 rows as 2 tiles) x (col group w>>2: 8 col tiles)
// ---------------------------------------------------------------------------
__global__ __launch_bounds__(256) void k_gemm1(
    const _Float16* __restrict__ tF, const float* __restrict__ aF,
    const _Float16* __restrict__ W1t, const float* __restrict__ b1,
    _Float16* __restrict__ y1) {
  __shared__ __align__(16) _Float16 sT[128 * T_PAD];   // 40 KB raw _t tile
  __shared__ __align__(16) _Float16 sA[128 * A_PAD];   // 8.5 KB scale tile
  const int r0 = blockIdx.y * 128;
  const int c0 = blockIdx.x * 256;
  const int tid = threadIdx.x;
  const int w = tid >> 5, lane = tid & 31;
  const int ln = lane & 15, half = lane >> 4;
  const int wr = w & 3, wc = w >> 2;

  // ---- stage _t tile (vector copy) ----
  {
    const int fn = tid >> 1;
    const int fc0 = (tid & 1) * 80;
    const _Float16* src = tF + (long long)(r0 + fn) * T_PAD + fc0;
    _Float16* dst = sT + fn * T_PAD + fc0;
#pragma unroll
    for (int j = 0; j < 10; ++j) *(v8h*)(dst + j * 8) = *(const v8h*)(src + j * 8);
  }
  // ---- stage _a tile as f16 scales ----
  for (int idx = tid; idx < 128 * A_COLS; idx += 256) {
    const int row = idx / A_COLS, c = idx % A_COLS;
    sA[row * A_PAD + c] = (_Float16)aF[(long long)(r0 + row) * A_COLS + c];
  }
  __syncthreads();

  v8f acc[2][8] = {};

  for (int ks = 0; ks < 5; ++ks) {
    // raw (unscaled) A fragments: loaded once per ks, reused for all 33 a's
    v16h araw[2];
#pragma unroll
    for (int rt = 0; rt < 2; ++rt) {
      const _Float16* ap = sT + (wr * 32 + rt * 16 + ln) * T_PAD + ks * 32 + 8 * half;
      araw[rt] = frag16(ap, ap + 16);
    }
    for (int a = 0; a < A_COLS; ++a) {
      const _Float16 s0 = sA[(wr * 32 + ln) * A_PAD + a];
      const _Float16 s1 = sA[(wr * 32 + 16 + ln) * A_PAD + a];
      v16h a0, a1;
#pragma unroll
      for (int e = 0; e < 16; ++e) {
        a0[e] = araw[0][e] * s0;
        a1[e] = araw[1][e] * s1;
      }
      const _Float16* Wa = W1t + (long long)a * 768 * T_PAD + ks * 32 + 8 * half;
#pragma unroll
      for (int ct = 0; ct < 8; ++ct) {
        const int p = c0 + wc * 128 + ct * 16 + ln;
        const _Float16* bp = Wa + (long long)p * T_PAD;
        v16h bf = frag16(bp, bp + 16);
        acc[0][ct] = wmma_f16(a0, bf, acc[0][ct]);
        acc[1][ct] = wmma_f16(a1, bf, acc[1][ct]);
      }
    }
  }

  // epilogue: bias + relu, store f16
#pragma unroll
  for (int ct = 0; ct < 8; ++ct) {
    const int p = c0 + wc * 128 + ct * 16 + ln;
    const float bias = b1[p];
#pragma unroll
    for (int rt = 0; rt < 2; ++rt) {
#pragma unroll
      for (int r = 0; r < 8; ++r) {
        const long long n = r0 + wr * 32 + rt * 16 + half * 8 + r;
        float v = acc[rt][ct][r] + bias;
        y1[n * 768 + p] = (_Float16)(v > 0.0f ? v : 0.0f);
      }
    }
  }
}

// ---------------------------------------------------------------------------
// K3: out = relu(y1 @ W2 + b2) + text_x. Operands straight from global
// (W2t = 1.2 MB and y1 tiles are L2-resident).
// ---------------------------------------------------------------------------
__global__ __launch_bounds__(256) void k_gemm2(
    const _Float16* __restrict__ y1, const _Float16* __restrict__ W2t,
    const float* __restrict__ b2, const float* __restrict__ text_x,
    float* __restrict__ out) {
  const int r0 = blockIdx.y * 128;
  const int c0 = blockIdx.x * 256;
  const int tid = threadIdx.x;
  const int w = tid >> 5, lane = tid & 31;
  const int ln = lane & 15, half = lane >> 4;
  const int wr = w & 3, wc = w >> 2;

  v8f acc[2][8] = {};

  for (int ks = 0; ks < 24; ++ks) {
    v16h af[2];
#pragma unroll
    for (int rt = 0; rt < 2; ++rt) {
      const long long n = r0 + wr * 32 + rt * 16 + ln;
      const _Float16* ap = y1 + n * 768 + ks * 32 + 8 * half;
      af[rt] = frag16(ap, ap + 16);
    }
#pragma unroll
    for (int ct = 0; ct < 8; ++ct) {
      const int p = c0 + wc * 128 + ct * 16 + ln;
      const _Float16* bp = W2t + (long long)p * 768 + ks * 32 + 8 * half;
      v16h bf = frag16(bp, bp + 16);
      acc[0][ct] = wmma_f16(af[0], bf, acc[0][ct]);
      acc[1][ct] = wmma_f16(af[1], bf, acc[1][ct]);
    }
  }

#pragma unroll
  for (int ct = 0; ct < 8; ++ct) {
    const int p = c0 + wc * 128 + ct * 16 + ln;
    const float bias = b2[p];
#pragma unroll
    for (int rt = 0; rt < 2; ++rt) {
#pragma unroll
      for (int r = 0; r < 8; ++r) {
        const long long n = r0 + wr * 32 + rt * 16 + half * 8 + r;
        float v = acc[rt][ct][r] + bias;
        v = v > 0.0f ? v : 0.0f;
        out[n * 768 + p] = v + text_x[n * 768 + p];
      }
    }
  }
}

// ---------------------------------------------------------------------------
// Host-side launcher
// ---------------------------------------------------------------------------
extern "C" void kernel_launch(void* const* d_in, const int* in_sizes, int n_in,
                              void* d_out, int out_size, void* d_ws, size_t ws_size,
                              hipStream_t stream) {
  (void)in_sizes; (void)n_in; (void)out_size; (void)ws_size;

  const float* text_x  = (const float*)d_in[0];
  const float* audio_x = (const float*)d_in[1];
  const float* a_gamma = (const float*)d_in[2];
  const float* a_beta  = (const float*)d_in[3];
  const float* a_mean  = (const float*)d_in[4];
  const float* a_var   = (const float*)d_in[5];
  const float* aW      = (const float*)d_in[6];
  const float* ab      = (const float*)d_in[7];
  const float* t_gamma = (const float*)d_in[8];
  const float* t_beta  = (const float*)d_in[9];
  const float* t_mean  = (const float*)d_in[10];
  const float* t_var   = (const float*)d_in[11];
  const float* tW      = (const float*)d_in[12];
  const float* tb      = (const float*)d_in[13];
  const float* W1      = (const float*)d_in[14];
  const float* b1      = (const float*)d_in[15];
  const float* W2      = (const float*)d_in[16];
  const float* b2      = (const float*)d_in[17];
  float* out = (float*)d_out;

  // workspace layout (all offsets 256B-aligned)
  char* ws = (char*)d_ws;
  size_t o = 0;
  _Float16* W1t  = (_Float16*)(ws + o); o += (size_t)NW1 * 2;       // 8,110,080
  _Float16* W2t  = (_Float16*)(ws + o); o += (size_t)NW2 * 2;       // 1,179,648
  _Float16* tWeT = (_Float16*)(ws + o); o += (size_t)NTW * 2;       //   196,608
  _Float16* aWeT = (_Float16*)(ws + o); o += (size_t)NAW * 2;       //     8,192
  float*    biasT = (float*)(ws + o);   o += 512;
  float*    biasA = (float*)(ws + o);   o += 256;
  _Float16* tFus = (_Float16*)(ws + o); o += (size_t)N_ROWS * T_PAD * 2;  // 5,242,880
  float*    aFus = (float*)(ws + o);    o += (size_t)N_ROWS * A_COLS * 4; // 2,162,688
  _Float16* y1ws = (_Float16*)(ws + o); o += (size_t)N_ROWS * 768 * 2;    // 25,165,824

  // K0: weight prep
  {
    const long long total = K0_TOTAL;
    const int blocks = (int)((total + 255) / 256);
    k_prep<<<blocks, 256, 0, stream>>>(aW, ab, a_gamma, a_beta, a_mean, a_var,
                                       tW, tb, t_gamma, t_beta, t_mean, t_var,
                                       W1, W2, W1t, W2t, tWeT, aWeT, biasT, biasA);
  }
  // K1: branch GEMMs -> _t, _a
  k_branches<<<N_ROWS / 128, 256, 0, stream>>>(text_x, audio_x, tWeT, biasT,
                                               aWeT, biasA, tFus, aFus);
  // K2: big fusion GEMM -> y1 (f16)
  k_gemm1<<<dim3(3, N_ROWS / 128), 256, 0, stream>>>(tFus, aFus, W1t, b1, y1ws);
  // K3: second GEMM + residual -> out (f32)
  k_gemm2<<<dim3(3, N_ROWS / 128), 256, 0, stream>>>(y1ws, W2t, b2, text_x, out);
}
